// GAT_53523882442952
// MI455X (gfx1250) — matrix operand order repaired and checked by reading, not compiled, not run on hardware
//
#include <hip/hip_runtime.h>

// ---------------------------------------------------------------------------
// Types for CDNA5 WMMA (wave32)
// ---------------------------------------------------------------------------
typedef __attribute__((ext_vector_type(16))) __bf16 v16bf;
typedef __attribute__((ext_vector_type(8)))  float  v8f;

union BF16x16 { v16bf v; uint4 q[2]; unsigned short s[16]; };

__device__ __forceinline__ unsigned short f2bf(float f) {
    unsigned int u = __float_as_uint(f);
    unsigned int r = u + 0x7FFFu + ((u >> 16) & 1u);   // round-to-nearest-even
    return (unsigned short)(r >> 16);
}

// Monotonic encoding of float for unsigned atomicMax
__device__ __forceinline__ unsigned int encf(float f) {
    unsigned int u = __float_as_uint(f);
    return (u & 0x80000000u) ? ~u : (u | 0x80000000u);
}
__device__ __forceinline__ float decf(unsigned int e) {
    unsigned int u = (e & 0x80000000u) ? (e & 0x7FFFFFFFu) : ~e;
    return __uint_as_float(u);
}

// ---------------------------------------------------------------------------
// f32 -> bf16 conversion (flat)
// ---------------------------------------------------------------------------
__global__ void cvt_bf16_kernel(const float* __restrict__ in,
                                unsigned short* __restrict__ out, long long n) {
    long long i = (long long)blockIdx.x * blockDim.x + threadIdx.x;
    if (i < n) out[i] = f2bf(in[i]);
}

// f32 [K,NC] -> bf16 [K,NCpad], zero-padded columns
__global__ void cvt_pad_bf16_kernel(const float* __restrict__ in,
                                    unsigned short* __restrict__ out,
                                    int K, int NC, int NCpad) {
    int i = blockIdx.x * blockDim.x + threadIdx.x;
    if (i >= K * NCpad) return;
    int k = i / NCpad, n = i % NCpad;
    out[i] = (n < NC) ? f2bf(in[k * NC + n]) : (unsigned short)0;
}

// ---------------------------------------------------------------------------
// WMMA GEMM: C[nrows,ncols] = A[nrows,K](bf16) * B[K,NCpad](bf16), f32 acc.
// Block = 256 threads = 8 wave32 (4 M x 2 N). Block tile 128(M) x 64(N).
// Wave tile 32x32 = 4 accumulators, 4 WMMA per K-step.
// LDS stride 40 ushorts (80B) => all fragment reads are 16B-aligned b128.
// B is staged transposed [n][k] so B-fragments are contiguous runs.
// ---------------------------------------------------------------------------
__global__ __launch_bounds__(256) void gemm_bf16_kernel(
    const unsigned short* __restrict__ A,
    const unsigned short* __restrict__ B,   // padded to NCpad columns
    float* __restrict__ C,
    int nrows, int K, int ncols, int NCpad)
{
    __shared__ unsigned short ldsA[128 * 40];   // [m][k], stride 40
    __shared__ unsigned short ldsBT[64 * 40];   // [n][k], stride 40

    const int tid  = threadIdx.x;
    const int lane = tid & 31;
    const int wid  = tid >> 5;     // 0..7
    const int wm   = wid & 3;      // wave M tile (0..3) -> rows wm*32..+31
    const int wn   = wid >> 2;     // wave N tile (0..1) -> cols wn*32..+31
    const int rowBase = blockIdx.x * 128;
    const int colBase = blockIdx.y * 64;

    v8f c00 = {}, c01 = {}, c10 = {}, c11 = {};

    const int mrow0 = wm * 32 + (lane & 15);        // A frag 0 row (local)
    const int mrow1 = mrow0 + 16;                   // A frag 1 row (local)
    const int ncol0 = wn * 32 + (lane & 15);        // B frag 0 col (local)
    const int ncol1 = ncol0 + 16;                   // B frag 1 col (local)
    const int kaddA = (lane >= 16) ? 8 : 0;
    const int kaddB = (lane >= 16) ? 8 : 0;

    for (int k0 = 0; k0 < K; k0 += 32) {
        __syncthreads();
        // --- stage A tile 128x32 bf16: 16 elems (2 x uint4) per thread ---
        {
            int r    = tid >> 1;            // 0..127
            int half = (tid & 1) * 16;      // 0 or 16
            int gr   = rowBase + r;
            uint4* q = (uint4*)&ldsA[r * 40 + half];
            if (gr < nrows) {
                const uint4* p = (const uint4*)(A + (size_t)gr * K + k0 + half);
                q[0] = p[0];
                q[1] = p[1];
            } else {
                uint4 z = {0u, 0u, 0u, 0u};
                q[0] = z;
                q[1] = z;
            }
        }
        // --- stage B tile 32(K) x 64(N), transposed into LDS [n][k] ---
        {
            int k  = tid >> 3;              // 0..31
            int n0 = (tid & 7) * 8;         // 0..56
            const uint4 d = *(const uint4*)(B + (size_t)(k0 + k) * NCpad + colBase + n0);
            const unsigned short* ds16 = (const unsigned short*)&d;
            #pragma unroll
            for (int i = 0; i < 8; ++i)
                ldsBT[(n0 + i) * 40 + k] = ds16[i];
        }
        __syncthreads();

        // --- fragments: contiguous 16B-aligned runs ---
        BF16x16 a0, a1, b0, b1;
        a0.q[0] = *(const uint4*)&ldsA[mrow0 * 40 + kaddA];
        a0.q[1] = *(const uint4*)&ldsA[mrow0 * 40 + 16 + kaddA];
        a1.q[0] = *(const uint4*)&ldsA[mrow1 * 40 + kaddA];
        a1.q[1] = *(const uint4*)&ldsA[mrow1 * 40 + 16 + kaddA];
        b0.q[0] = *(const uint4*)&ldsBT[ncol0 * 40 + kaddB];
        b0.q[1] = *(const uint4*)&ldsBT[ncol0 * 40 + kaddB + 16];
        b1.q[0] = *(const uint4*)&ldsBT[ncol1 * 40 + kaddB];
        b1.q[1] = *(const uint4*)&ldsBT[ncol1 * 40 + kaddB + 16];

        c00 = __builtin_amdgcn_wmma_f32_16x16x32_bf16(false, a0.v, false, b0.v, (short)0, c00, false, false);
        c01 = __builtin_amdgcn_wmma_f32_16x16x32_bf16(false, a0.v, false, b1.v, (short)0, c01, false, false);
        c10 = __builtin_amdgcn_wmma_f32_16x16x32_bf16(false, a1.v, false, b0.v, (short)0, c10, false, false);
        c11 = __builtin_amdgcn_wmma_f32_16x16x32_bf16(false, a1.v, false, b1.v, (short)0, c11, false, false);
    }

    // --- store C (16x16 f32 layout: VGPR r -> M = r + (lane>=16)*8) ---
    const int madd = (lane >= 16) ? 8 : 0;
    const int nA = colBase + wn * 32 + (lane & 15);
    const int nB = nA + 16;
    #pragma unroll
    for (int r = 0; r < 8; ++r) {
        int m0 = rowBase + wm * 32 + r + madd;
        int m1 = m0 + 16;
        if (m0 < nrows) {
            if (nA < ncols) C[(size_t)m0 * ncols + nA] = c00[r];
            if (nB < ncols) C[(size_t)m0 * ncols + nB] = c01[r];
        }
        if (m1 < nrows) {
            if (nA < ncols) C[(size_t)m1 * ncols + nA] = c10[r];
            if (nB < ncols) C[(size_t)m1 * ncols + nB] = c11[r];
        }
    }
}

// ---------------------------------------------------------------------------
// Per-node attention logits: as[n,h] = sum_c h[n,h,c]*a_src[h,c], same for ad
// ---------------------------------------------------------------------------
__global__ void alpha_kernel(const float* __restrict__ h,
                             const float* __restrict__ a_s,
                             const float* __restrict__ a_d,
                             float* __restrict__ as_, float* __restrict__ ad_,
                             int n_nodes, int C)
{
    int i = blockIdx.x * blockDim.x + threadIdx.x;   // over n*3
    if (i >= n_nodes * 3) return;
    int n = i / 3, hh = i % 3;
    const float* row = h + (size_t)n * (3 * C) + hh * C;
    const float* ws  = a_s + hh * C;
    const float* wd  = a_d + hh * C;
    float s = 0.f, d = 0.f;
    for (int c = 0; c < C; ++c) { s += row[c] * ws[c]; d += row[c] * wd[c]; }
    as_[i] = s; ad_[i] = d;
}

// ---------------------------------------------------------------------------
// Edge pass 1: leaky-relu score, store, atomicMax (encoded) per (dst, head)
// ---------------------------------------------------------------------------
__global__ void edge_score_max_kernel(const int* __restrict__ ei, int E_, int n_nodes,
                                      const float* __restrict__ as_,
                                      const float* __restrict__ ad_,
                                      float* __restrict__ sbuf,
                                      unsigned int* __restrict__ emax)
{
    long long total = (long long)(E_ + n_nodes) * 3;
    for (long long i = (long long)blockIdx.x * blockDim.x + threadIdx.x;
         i < total; i += (long long)gridDim.x * blockDim.x) {
        long long e = i / 3; int hh = (int)(i % 3);
        int s, d;
        if (e < E_) { s = ei[e]; d = ei[E_ + e]; }
        else        { s = d = (int)(e - E_); }
        float sc = as_[(size_t)s * 3 + hh] + ad_[(size_t)d * 3 + hh];
        float lr = sc > 0.f ? sc : 0.2f * sc;
        sbuf[i] = lr;
        atomicMax(&emax[(size_t)d * 3 + hh], encf(lr));
    }
}

// ---------------------------------------------------------------------------
// Edge pass 2: w = exp(score - max[dst]); store w; denom[dst,h] += w
// ---------------------------------------------------------------------------
__global__ void edge_exp_denom_kernel(const int* __restrict__ ei, int E_, int n_nodes,
                                      const unsigned int* __restrict__ emax,
                                      float* __restrict__ sbuf,
                                      float* __restrict__ denom)
{
    long long total = (long long)(E_ + n_nodes) * 3;
    for (long long i = (long long)blockIdx.x * blockDim.x + threadIdx.x;
         i < total; i += (long long)gridDim.x * blockDim.x) {
        long long e = i / 3; int hh = (int)(i % 3);
        int d = (e < E_) ? ei[E_ + e] : (int)(e - E_);
        float w = __expf(sbuf[i] - decf(emax[(size_t)d * 3 + hh]));
        sbuf[i] = w;
        atomicAdd(&denom[(size_t)d * 3 + hh], w);
    }
}

// ---------------------------------------------------------------------------
// Edge pass 3: acc[dst, h, c] += w(e,h) * hfeat[src, h, c]   (c fastest)
// ---------------------------------------------------------------------------
__global__ void edge_aggregate_kernel(const int* __restrict__ ei, int E_, int n_nodes,
                                      int C,
                                      const float* __restrict__ sbuf,
                                      const float* __restrict__ h,
                                      float* __restrict__ acc)
{
    const int HC = 3 * C;
    long long total = (long long)(E_ + n_nodes) * HC;
    for (long long i = (long long)blockIdx.x * blockDim.x + threadIdx.x;
         i < total; i += (long long)gridDim.x * blockDim.x) {
        long long e = i / HC; int j = (int)(i % HC); int hh = j / C;
        int s, d;
        if (e < E_) { s = ei[e]; d = ei[E_ + e]; }
        else        { s = d = (int)(e - E_); }
        float w = sbuf[e * 3 + hh];
        atomicAdd(&acc[(size_t)d * HC + j], w * h[(size_t)s * HC + j]);
    }
}

// ---------------------------------------------------------------------------
// Finalize (layers 1,2): v = acc/denom + b ; ELU ; emit bf16 for next GEMM
// ---------------------------------------------------------------------------
__global__ void finalize_elu_kernel(const float* __restrict__ acc,
                                    const float* __restrict__ denom,
                                    const float* __restrict__ bias,
                                    unsigned short* __restrict__ actbf,
                                    int n_nodes, int C)
{
    const int HC = 3 * C;
    long long total = (long long)n_nodes * HC;
    long long i = (long long)blockIdx.x * blockDim.x + threadIdx.x;
    if (i >= total) return;
    int n = (int)(i / HC), j = (int)(i % HC), hh = j / C;
    float v = acc[i] / (denom[(size_t)n * 3 + hh] + 1e-16f) + bias[j];
    v = v > 0.f ? v : (__expf(v) - 1.f);     // ELU(alpha=1)
    actbf[i] = f2bf(v);
}

// ---------------------------------------------------------------------------
// Finalize layer 3 + mean-pool scatter: pooled[batch[n], j] += v
// ---------------------------------------------------------------------------
__global__ void finalize_pool_kernel(const float* __restrict__ acc,
                                     const float* __restrict__ denom,
                                     const float* __restrict__ bias,
                                     const int* __restrict__ batch,
                                     float* __restrict__ pooled,
                                     int n_nodes, int C)
{
    const int HC = 3 * C;   // 48
    long long total = (long long)n_nodes * HC;
    long long i = (long long)blockIdx.x * blockDim.x + threadIdx.x;
    if (i >= total) return;
    int n = (int)(i / HC), j = (int)(i % HC), hh = j / C;
    float v = acc[i] / (denom[(size_t)n * 3 + hh] + 1e-16f) + bias[j];
    atomicAdd(&pooled[(size_t)batch[n] * HC + j], v);
}

__global__ void count_kernel(const int* __restrict__ batch,
                             float* __restrict__ counts, int n_nodes)
{
    int n = blockIdx.x * blockDim.x + threadIdx.x;
    if (n < n_nodes) atomicAdd(&counts[batch[n]], 1.f);
}

// ---------------------------------------------------------------------------
// Head: mean-pool finish, ReLU, 48->1 linear, sigmoid out, BCE loss term
// ---------------------------------------------------------------------------
__global__ void head_kernel(const float* __restrict__ pooled,
                            const float* __restrict__ counts,
                            const float* __restrict__ Wl,
                            const float* __restrict__ bl,
                            const float* __restrict__ y,
                            float* __restrict__ out,
                            float* __restrict__ lossAcc, int Gn)
{
    int g = blockIdx.x * blockDim.x + threadIdx.x;
    if (g >= Gn) return;
    float cnt = counts[g]; if (cnt < 1.f) cnt = 1.f;
    float logit = bl[0];
    #pragma unroll
    for (int j = 0; j < 48; ++j) {
        float p = pooled[(size_t)g * 48 + j] / cnt;
        p = p > 0.f ? p : 0.f;
        logit += p * Wl[j];
    }
    out[g] = 1.f / (1.f + __expf(-logit));
    // stable softplus
    float sp = (logit > 0.f ? logit : 0.f) + log1pf(__expf(-fabsf(logit)));
    atomicAdd(lossAcc, sp - y[g] * logit);
}

__global__ void loss_finish_kernel(const float* __restrict__ lossAcc,
                                   float* __restrict__ out, int Gn)
{
    if (blockIdx.x == 0 && threadIdx.x == 0) out[Gn] = lossAcc[0] / (float)Gn;
}

// ---------------------------------------------------------------------------
// Host side
// ---------------------------------------------------------------------------
static inline size_t alignUp256(size_t x) { return (x + 255) & ~(size_t)255; }

extern "C" void kernel_launch(void* const* d_in, const int* in_sizes, int n_in,
                              void* d_out, int out_size, void* d_ws, size_t ws_size,
                              hipStream_t stream)
{
    (void)n_in; (void)out_size; (void)ws_size;
    const float* x   = (const float*)d_in[0];
    const float* y   = (const float*)d_in[1];
    const int*   ei  = (const int*)d_in[2];
    const int*   bat = (const int*)d_in[3];
    const float* W1  = (const float*)d_in[4];
    const float* as1 = (const float*)d_in[5];
    const float* ad1 = (const float*)d_in[6];
    const float* b1  = (const float*)d_in[7];
    const float* W2  = (const float*)d_in[8];
    const float* as2 = (const float*)d_in[9];
    const float* ad2 = (const float*)d_in[10];
    const float* b2  = (const float*)d_in[11];
    const float* W3  = (const float*)d_in[12];
    const float* as3 = (const float*)d_in[13];
    const float* ad3 = (const float*)d_in[14];
    const float* b3  = (const float*)d_in[15];
    const float* Wl  = (const float*)d_in[16];
    const float* bl  = (const float*)d_in[17];
    float* out = (float*)d_out;

    const int FIN = 128;
    const int Nn  = in_sizes[0] / FIN;     // 100000
    const int Gn  = in_sizes[1];           // 5000
    const int Ee  = in_sizes[2] / 2;       // 1600000
    const int H3  = 3;
    const int K1 = 128, NC1 = 192, NP1 = 192;   // NCpad = round-up to 64
    const int K2 = 192, NC2 = 96,  NP2 = 128;
    const int K3 = 96,  NC3 = 48,  NP3 = 64;

    // workspace layout
    char* ws = (char*)d_ws;
    size_t o = 0;
    auto take = [&](size_t bytes) { size_t r = o; o = alignUp256(o + bytes); return r; };
    unsigned short* w1b   = (unsigned short*)(ws + take((size_t)K1 * NP1 * 2));
    unsigned short* w2b   = (unsigned short*)(ws + take((size_t)K2 * NP2 * 2));
    unsigned short* w3b   = (unsigned short*)(ws + take((size_t)K3 * NP3 * 2));
    unsigned short* actbf = (unsigned short*)(ws + take((size_t)Nn * 192 * 2));
    float*        hfeat = (float*)(ws + take((size_t)Nn * 192 * 4));
    float*        accb  = (float*)(ws + take((size_t)Nn * 192 * 4));
    float*        asb   = (float*)(ws + take((size_t)Nn * H3 * 4));
    float*        adb   = (float*)(ws + take((size_t)Nn * H3 * 4));
    unsigned int* emaxb = (unsigned int*)(ws + take((size_t)Nn * H3 * 4));
    float*        denb  = (float*)(ws + take((size_t)Nn * H3 * 4));
    float*        sbuf  = (float*)(ws + take((size_t)(Ee + Nn) * H3 * 4));
    float*        pooled= (float*)(ws + take((size_t)Gn * 48 * 4));
    float*        counts= (float*)(ws + take((size_t)Gn * 4));
    float*        lossA = (float*)(ws + take(256));

    const int T = 256;
    auto blk = [](long long n, int t) { return (int)((n + t - 1) / t); };

    // --- convert weights (padded) + x to bf16 ---
    cvt_pad_bf16_kernel<<<blk((long long)K1 * NP1, T), T, 0, stream>>>(W1, w1b, K1, NC1, NP1);
    cvt_pad_bf16_kernel<<<blk((long long)K2 * NP2, T), T, 0, stream>>>(W2, w2b, K2, NC2, NP2);
    cvt_pad_bf16_kernel<<<blk((long long)K3 * NP3, T), T, 0, stream>>>(W3, w3b, K3, NC3, NP3);
    cvt_bf16_kernel<<<blk((long long)Nn * FIN, T), T, 0, stream>>>(x, actbf, (long long)Nn * FIN);

    long long eTot = (long long)(Ee + Nn) * H3;
    int eBlocks = blk(eTot, T);

    struct Layer {
        const unsigned short* Wb; int K, NC, NP, C;
        const float *asv, *adv, *bias;
    } layers[3] = {
        { w1b, K1, NC1, NP1, 64, as1, ad1, b1 },
        { w2b, K2, NC2, NP2, 32, as2, ad2, b2 },
        { w3b, K3, NC3, NP3, 16, as3, ad3, b3 },
    };

    for (int L = 0; L < 3; ++L) {
        const Layer& ly = layers[L];
        // zero accumulators
        hipMemsetAsync(accb,  0, (size_t)Nn * ly.NC * 4, stream);
        hipMemsetAsync(emaxb, 0, (size_t)Nn * H3 * 4, stream);   // 0 < any encf(float)
        hipMemsetAsync(denb,  0, (size_t)Nn * H3 * 4, stream);

        // GEMM: hfeat = actbf @ Wb
        dim3 gg(blk(Nn, 128), ly.NP / 64);
        gemm_bf16_kernel<<<gg, 256, 0, stream>>>(actbf, ly.Wb, hfeat, Nn, ly.K, ly.NC, ly.NP);

        // attention logits
        alpha_kernel<<<blk((long long)Nn * H3, T), T, 0, stream>>>(
            hfeat, ly.asv, ly.adv, asb, adb, Nn, ly.C);

        // edge passes
        edge_score_max_kernel<<<eBlocks, T, 0, stream>>>(ei, Ee, Nn, asb, adb, sbuf, emaxb);
        edge_exp_denom_kernel<<<eBlocks, T, 0, stream>>>(ei, Ee, Nn, emaxb, sbuf, denb);

        long long aTot = (long long)(Ee + Nn) * (3 * ly.C);
        long long ab = (aTot + T - 1) / T;
        if (ab > 262144) ab = 262144;
        edge_aggregate_kernel<<<(int)ab, T, 0, stream>>>(ei, Ee, Nn, ly.C, sbuf, hfeat, accb);

        if (L < 2) {
            finalize_elu_kernel<<<blk((long long)Nn * ly.NC, T), T, 0, stream>>>(
                accb, denb, ly.bias, actbf, Nn, ly.C);
        } else {
            hipMemsetAsync(pooled, 0, (size_t)Gn * 48 * 4, stream);
            hipMemsetAsync(counts, 0, (size_t)Gn * 4, stream);
            hipMemsetAsync(lossA,  0, 4, stream);
            finalize_pool_kernel<<<blk((long long)Nn * ly.NC, T), T, 0, stream>>>(
                accb, denb, ly.bias, bat, pooled, Nn, ly.C);
        }
    }

    count_kernel<<<blk(Nn, T), T, 0, stream>>>(bat, counts, Nn);
    head_kernel<<<blk(Gn, T), T, 0, stream>>>(pooled, counts, Wl, bl, y, out, lossA, Gn);
    loss_finish_kernel<<<1, 1, 0, stream>>>(lossA, out, Gn);
}
